// Attention_73169062855367
// MI455X (gfx1250) — compile-verified, matrix operand order checked by reading, not run on hardware
//
#include <hip/hip_runtime.h>
#include <hip/hip_bf16.h>
#include <math.h>

typedef __attribute__((ext_vector_type(16))) _Float16 v16h;
typedef __attribute__((ext_vector_type(8)))  _Float16 v8h;
typedef __attribute__((ext_vector_type(8)))  float    v8f;
typedef __attribute__((ext_vector_type(4)))  unsigned int u32x4;
typedef __attribute__((ext_vector_type(8)))  int      i32x8;
typedef __attribute__((ext_vector_type(4)))  int      i32x4;

#define DIM   1024
#define NTOK  2048
#define BATCH 2
#define HEADS 16
#define HD    64
#define ROWS  (BATCH * NTOK)   // 4096
#define SCALE 0.03125f         // dim^-0.5 = 1/32 (module scales by dim, not head_dim)

// ---------------------------------------------------------------------------
// A-operand loader for v_wmma_f32_16x16x32_f16 (wave32), works on LDS or
// global pointers. 16x32 f16 tile, row-major, leading dim `ld` (elements).
// ---------------------------------------------------------------------------
static __device__ inline v16h load_a_tile(const _Float16* __restrict__ src, int ld) {
  int lane = threadIdx.x & 31;
  int m    = lane & 15;
  int koff = (lane >> 4) << 3;  // 0 or 8
  const _Float16* p = src + m * ld + koff;
  v8h lo = *(const v8h*)(p);
  v8h hi = *(const v8h*)(p + 16);
  v16h r;
#pragma unroll
  for (int i = 0; i < 8; ++i) { r[i] = lo[i]; r[i + 8] = hi[i]; }
  return r;
}

static __device__ inline v8f wmma_f16(v16h a, v16h b, v8f c) {
  return __builtin_amdgcn_wmma_f32_16x16x32_f16(
      false, a, false, b, (short)0, c, false, false);
}

// LDS byte offset of a __shared__ object: flat addr = aperture‖offset,
// low 32 bits are the workgroup-relative LDS byte address.
static __device__ inline unsigned int lds_off(const void* p) {
  return (unsigned int)(unsigned long long)(uintptr_t)p;
}

// ---------------------------------------------------------------------------
// TDM: async 2D tile load global -> LDS (tensor_load_to_lds).
// D# per ISA 8.3/8.4: 2-byte elements, 2-D tensor (groups 2/3 zero).
// All fields uniform (SGPR-resident). Tracked by TENSORcnt.
// ---------------------------------------------------------------------------
static __device__ inline void tdm_load_2d(unsigned int lds_addr, const void* gptr,
                                          unsigned int tile_w, unsigned int tile_h,
                                          unsigned int tensor_w, unsigned int tensor_h,
                                          unsigned int stride_elems) {
  unsigned long long ga = (unsigned long long)(uintptr_t)gptr;
  u32x4 g0;
  g0[0] = 1u;                                        // count=1, user mode
  g0[1] = lds_addr;                                  // lds_addr [63:32]
  g0[2] = (unsigned int)(ga & 0xffffffffu);          // global_addr lo
  g0[3] = (unsigned int)((ga >> 32) & 0x1ffffffu) | (2u << 30);  // addr hi | type=2
  i32x8 g1;
  g1[0] = (int)(1u << 16);                           // wg_mask=0, data_size=1 (2B)
  g1[1] = (int)((tensor_w & 0xffffu) << 16);         // [63:48] tensor_dim0 lo16
  g1[2] = (int)(((tensor_w >> 16) & 0xffffu) |       // [79:64] tensor_dim0 hi16
                ((tensor_h & 0xffffu) << 16));       // [95:80] tensor_dim1 lo16
  g1[3] = (int)(((tensor_h >> 16) & 0xffffu) |       // [111:96] tensor_dim1 hi16
                ((tile_w & 0xffffu) << 16));         // [127:112] tile_dim0
  g1[4] = (int)(tile_h & 0xffffu);                   // [143:128] tile_dim1; tile_dim2=0
  g1[5] = (int)stride_elems;                         // [191:160] tensor_dim0_stride lo32
  g1[6] = 0;
  g1[7] = 0;
  i32x4 z4 = {0, 0, 0, 0};
  i32x8 z8 = {0, 0, 0, 0, 0, 0, 0, 0};
  __builtin_amdgcn_tensor_load_to_lds(g0, g1, z4, z4, z8, 0);
}

// ---------------------------------------------------------------------------
// Prep: fp32 -> f16 convert, and fp32 -> f16 transpose for weights.
// ---------------------------------------------------------------------------
__global__ void prep_cvt(const float* __restrict__ x, _Float16* __restrict__ xh, int n) {
  int i = blockIdx.x * blockDim.x + threadIdx.x;
  if (i < n) xh[i] = (_Float16)x[i];
}

__global__ void prep_transpose(const float* __restrict__ w, _Float16* __restrict__ wt,
                               int rows, int cols) {
  // wt[c*rows + k] = w[k*cols + c]
  int i = blockIdx.x * blockDim.x + threadIdx.x;
  if (i < rows * cols) {
    int c = i / rows;
    int k = i - c * rows;
    wt[i] = (_Float16)w[k * cols + c];
  }
}

// ---------------------------------------------------------------------------
// GEMM main loop, TDM double-buffered. One wave computes a 32x64 tile.
// Per 64-deep k-panel: TDM prefetches the next A(32x64)/B(64x64) panels into
// the alternate LDS buffers while 16 WMMAs consume the current ones from LDS.
// Outer loop pinned (no spills); per half-k all 6 tiles load into distinct
// registers before the 8 WMMAs so s_wait_dscnt can stagger.
// ---------------------------------------------------------------------------
static __device__ inline void gemm_loop_32x64_tdm(const _Float16* __restrict__ arow,
                                                  const _Float16* __restrict__ brow,
                                                  _Float16* lds_a /*[2][32*64]*/,
                                                  _Float16* lds_b /*[2][64*64]*/,
                                                  v8f acc[2][4]) {
  // prologue: panel 0 into buffer 0
  tdm_load_2d(lds_off(lds_a), arow, 64, 32, 64, 32, DIM);
  tdm_load_2d(lds_off(lds_b), brow, 64, 64, 64, 64, DIM);

  int buf = 0;
#pragma unroll 1
  for (int k0 = 0; k0 < DIM; k0 += 64, buf ^= 1) {
    if (k0 + 64 < DIM) {   // prefetch next panel into alternate buffer
      tdm_load_2d(lds_off(lds_a + (buf ^ 1) * 32 * 64), arow + k0 + 64,
                  64, 32, (unsigned)(DIM - k0 - 64), 32, DIM);
      tdm_load_2d(lds_off(lds_b + (buf ^ 1) * 64 * 64), brow + k0 + 64,
                  64, 64, (unsigned)(DIM - k0 - 64), 64, DIM);
      __builtin_amdgcn_s_wait_tensorcnt(2);  // current panel's 2 ops retired
    } else {
      __builtin_amdgcn_s_wait_tensorcnt(0);
    }
    const _Float16* A = lds_a + buf * 32 * 64;
    const _Float16* B = lds_b + buf * 64 * 64;
#pragma unroll
    for (int kc = 0; kc < 2; ++kc) {
      // issue all 6 tile loads (12 ds_load_b128) before any WMMA
      v16h a0 = load_a_tile(A + 0 * 16 * 64 + kc * 32, 64);
      v16h a1 = load_a_tile(A + 1 * 16 * 64 + kc * 32, 64);
      v16h b0 = load_a_tile(B + 0 * 16 * 64 + kc * 32, 64);
      v16h b1 = load_a_tile(B + 1 * 16 * 64 + kc * 32, 64);
      v16h b2 = load_a_tile(B + 2 * 16 * 64 + kc * 32, 64);
      v16h b3 = load_a_tile(B + 3 * 16 * 64 + kc * 32, 64);
      acc[0][0] = wmma_f16(a0, b0, acc[0][0]);
      acc[1][0] = wmma_f16(a1, b0, acc[1][0]);
      acc[0][1] = wmma_f16(a0, b1, acc[0][1]);
      acc[1][1] = wmma_f16(a1, b1, acc[1][1]);
      acc[0][2] = wmma_f16(a0, b2, acc[0][2]);
      acc[1][2] = wmma_f16(a1, b2, acc[1][2]);
      acc[0][3] = wmma_f16(a0, b3, acc[0][3]);
      acc[1][3] = wmma_f16(a1, b3, acc[1][3]);
    }
  }
}

// ---------------------------------------------------------------------------
// Stage 1: qkv = x @ w_qkv + b_qkv -> Q,K [b][h][n][64] f16; Vt [b][h][64][n]
// ---------------------------------------------------------------------------
__global__ void gemm_qkv(const _Float16* __restrict__ xh, const _Float16* __restrict__ wt,
                         const float* __restrict__ bias,
                         _Float16* __restrict__ q, _Float16* __restrict__ k,
                         _Float16* __restrict__ vt) {
  __shared__ __align__(16) _Float16 lds_a[2][32 * 64];
  __shared__ __align__(16) _Float16 lds_b[2][64 * 64];
  int mt = blockIdx.x;   // 0..127 (row tiles of 32)
  int nt = blockIdx.y;   // 0..47  (col tiles of 64)
  int lane = threadIdx.x & 31;

  v8f acc[2][4] = {};
  gemm_loop_32x64_tdm(xh + (size_t)mt * 32 * DIM,
                      wt + (size_t)nt * 64 * DIM,
                      &lds_a[0][0], &lds_b[0][0], acc);

  int nn    = lane & 15;
  int mbase = (lane >> 4) << 3;
#pragma unroll
  for (int t = 0; t < 4; ++t) {
    int col   = nt * 64 + t * 16 + nn;
    float bv  = bias[col];
    int which = col >> 10;        // 0=Q 1=K 2=V
    int h     = (col >> 6) & 15;
    int d     = col & 63;
#pragma unroll
    for (int m = 0; m < 2; ++m) {
#pragma unroll
      for (int r = 0; r < 8; ++r) {
        int row = mt * 32 + m * 16 + mbase + r;   // 0..4095
        int bb  = row >> 11;
        int sn  = row & (NTOK - 1);
        _Float16 val = (_Float16)(acc[m][t][r] + bv);
        if (which == 0)      q [(((size_t)(bb * HEADS + h) * NTOK) + sn) * HD + d]  = val;
        else if (which == 1) k [(((size_t)(bb * HEADS + h) * NTOK) + sn) * HD + d]  = val;
        else                 vt[(((size_t)(bb * HEADS + h) * HD)  + d) * NTOK + sn] = val;
      }
    }
  }
}

// ---------------------------------------------------------------------------
// Stage 2: flash attention, one wave per (b,h, 32-row tile). K (32x64) and
// Vt (64x32) panels stream through TDM-filled double-buffered LDS; online
// softmax in f32; P transposes C->A layout via LDS.
// ---------------------------------------------------------------------------
__global__ void flash_attn(const _Float16* __restrict__ q, const _Float16* __restrict__ kmat,
                           const _Float16* __restrict__ vt, _Float16* __restrict__ oh) {
  __shared__ __align__(16) _Float16 lds_k[2][32 * HD];   // 32 rows x 64
  __shared__ __align__(16) _Float16 lds_v[2][HD * 32];   // 64 rows x 32
  __shared__ __align__(16) _Float16 lds_p[32 * 32];

  int mt   = blockIdx.x;   // 0..63 row tile (32 rows)
  int bh   = blockIdx.y;   // 0..31 (b*16+h)
  int lane = threadIdx.x & 31;

  const _Float16* qb = q    + ((size_t)bh * NTOK + mt * 32) * HD;
  const _Float16* kb = kmat + (size_t)bh * NTOK * HD;
  const _Float16* vb = vt   + (size_t)bh * HD * NTOK;

  // prologue: panel 0
  tdm_load_2d(lds_off(lds_k[0]), kb, 64, 32, 64, NTOK, 64);
  tdm_load_2d(lds_off(lds_v[0]), vb, 32, 64, NTOK, 64, NTOK);

  // Preload Q tile (32 x 64) as four A operands
  v16h qa[2][2];
#pragma unroll
  for (int m = 0; m < 2; ++m) {
    qa[m][0] = load_a_tile(qb + (size_t)m * 16 * HD, HD);
    qa[m][1] = load_a_tile(qb + (size_t)m * 16 * HD + 32, HD);
  }

  float mrun[2][8], lrun[2][8];
  v8f o[2][4] = {};
#pragma unroll
  for (int m = 0; m < 2; ++m)
#pragma unroll
    for (int r = 0; r < 8; ++r) { mrun[m][r] = -INFINITY; lrun[m][r] = 0.f; }

  int nn    = lane & 15;
  int mbase = (lane >> 4) << 3;

  int buf = 0;
#pragma unroll 1
  for (int j0 = 0; j0 < NTOK; j0 += 32, buf ^= 1) {
    if (j0 + 32 < NTOK) {  // prefetch next K/V panel into alternate buffer
      tdm_load_2d(lds_off(lds_k[buf ^ 1]), kb + (size_t)(j0 + 32) * HD,
                  64, 32, 64, (unsigned)(NTOK - j0 - 32), 64);
      tdm_load_2d(lds_off(lds_v[buf ^ 1]), vb + (j0 + 32),
                  32, 64, (unsigned)(NTOK - j0 - 32), 64, NTOK);
      __builtin_amdgcn_s_wait_tensorcnt(2);
    } else {
      __builtin_amdgcn_s_wait_tensorcnt(0);
    }

    // ---- scores: S[32 x 32] = Q . K^T from LDS ----
    const _Float16* KP = lds_k[buf];
    v16h kt00 = load_a_tile(KP + 0 * 16 * 64 + 0,  64);
    v16h kt01 = load_a_tile(KP + 0 * 16 * 64 + 32, 64);
    v16h kt10 = load_a_tile(KP + 1 * 16 * 64 + 0,  64);
    v16h kt11 = load_a_tile(KP + 1 * 16 * 64 + 32, 64);
    v8f s[2][2] = {};
#pragma unroll
    for (int m = 0; m < 2; ++m) {
      s[m][0] = wmma_f16(qa[m][0], kt00, s[m][0]);
      s[m][0] = wmma_f16(qa[m][1], kt01, s[m][0]);
      s[m][1] = wmma_f16(qa[m][0], kt10, s[m][1]);
      s[m][1] = wmma_f16(qa[m][1], kt11, s[m][1]);
    }

    // ---- V tiles from LDS (overlap with softmax VALU work) ----
    const _Float16* VP = lds_v[buf];
    v16h vt0 = load_a_tile(VP + 0  * 32, 32);
    v16h vt1 = load_a_tile(VP + 16 * 32, 32);
    v16h vt2 = load_a_tile(VP + 32 * 32, 32);
    v16h vt3 = load_a_tile(VP + 48 * 32, 32);

    // ---- online softmax per 16-row group ----
    __syncthreads();   // LDS WAR vs previous iteration's P reads
#pragma unroll
    for (int m = 0; m < 2; ++m) {
      float p0[8], p1[8];
#pragma unroll
      for (int r = 0; r < 8; ++r) {
        float a0 = s[m][0][r] * SCALE;
        float a1 = s[m][1][r] * SCALE;
        float t = fmaxf(a0, a1);
        t = fmaxf(t, __shfl_xor(t, 1, 32));
        t = fmaxf(t, __shfl_xor(t, 2, 32));
        t = fmaxf(t, __shfl_xor(t, 4, 32));
        t = fmaxf(t, __shfl_xor(t, 8, 32));
        float mnew  = fmaxf(mrun[m][r], t);
        float alpha = __expf(mrun[m][r] - mnew);
        p0[r] = __expf(a0 - mnew);
        p1[r] = __expf(a1 - mnew);
        float rs = p0[r] + p1[r];
        rs += __shfl_xor(rs, 1, 32);
        rs += __shfl_xor(rs, 2, 32);
        rs += __shfl_xor(rs, 4, 32);
        rs += __shfl_xor(rs, 8, 32);
        lrun[m][r] = lrun[m][r] * alpha + rs;
        mrun[m][r] = mnew;
#pragma unroll
        for (int t4 = 0; t4 < 4; ++t4) o[m][t4][r] *= alpha;
      }
#pragma unroll
      for (int r = 0; r < 8; ++r) {
        lds_p[(m * 16 + mbase + r) * 32 + nn]      = (_Float16)p0[r];
        lds_p[(m * 16 + mbase + r) * 32 + 16 + nn] = (_Float16)p1[r];
      }
    }
    __syncthreads();

    // ---- O (32x64) += P (32x32) . V (32x64) ----
#pragma unroll
    for (int m = 0; m < 2; ++m) {
      v16h pa = load_a_tile(lds_p + m * 16 * 32, 32);
      o[m][0] = wmma_f16(pa, vt0, o[m][0]);
      o[m][1] = wmma_f16(pa, vt1, o[m][1]);
      o[m][2] = wmma_f16(pa, vt2, o[m][2]);
      o[m][3] = wmma_f16(pa, vt3, o[m][3]);
    }
  }

  // normalize; store O as f16 in [b*n][dim] (head-merged) for the out proj
  int b = bh >> 4, h = bh & 15;
#pragma unroll
  for (int m = 0; m < 2; ++m)
#pragma unroll
    for (int t4 = 0; t4 < 4; ++t4)
#pragma unroll
      for (int r = 0; r < 8; ++r) {
        int row   = mt * 32 + m * 16 + mbase + r;
        float val = o[m][t4][r] / lrun[m][r];
        oh[((size_t)(b * NTOK + row)) * DIM + h * HD + t4 * 16 + nn] = (_Float16)val;
      }
}

// ---------------------------------------------------------------------------
// Stage 3: out = O @ w_out + b_out (fp32 output)
// ---------------------------------------------------------------------------
__global__ void gemm_out(const _Float16* __restrict__ oh, const _Float16* __restrict__ wt,
                         const float* __restrict__ bias, float* __restrict__ out) {
  __shared__ __align__(16) _Float16 lds_a[2][32 * 64];
  __shared__ __align__(16) _Float16 lds_b[2][64 * 64];
  int mt = blockIdx.x;   // 0..127
  int nt = blockIdx.y;   // 0..15
  int lane = threadIdx.x & 31;

  v8f acc[2][4] = {};
  gemm_loop_32x64_tdm(oh + (size_t)mt * 32 * DIM,
                      wt + (size_t)nt * 64 * DIM,
                      &lds_a[0][0], &lds_b[0][0], acc);

  int nn    = lane & 15;
  int mbase = (lane >> 4) << 3;
#pragma unroll
  for (int t = 0; t < 4; ++t) {
    int col  = nt * 64 + t * 16 + nn;
    float bv = bias[col];
#pragma unroll
    for (int m = 0; m < 2; ++m)
#pragma unroll
      for (int r = 0; r < 8; ++r) {
        int row = mt * 32 + m * 16 + mbase + r;
        out[(size_t)row * DIM + col] = acc[m][t][r] + bv;
      }
  }
}

// ---------------------------------------------------------------------------
extern "C" void kernel_launch(void* const* d_in, const int* in_sizes, int n_in,
                              void* d_out, int out_size, void* d_ws, size_t ws_size,
                              hipStream_t stream) {
  const float* x     = (const float*)d_in[0];  // [2,2048,1024]
  const float* w_qkv = (const float*)d_in[1];  // [1024,3072]
  const float* b_qkv = (const float*)d_in[2];  // [3072]
  const float* w_out = (const float*)d_in[3];  // [1024,1024]
  const float* b_out = (const float*)d_in[4];  // [1024]
  float* out = (float*)d_out;                  // [2,2048,1024] fp32

  // workspace layout (all f16): ~48 MB total, fits L2 (192 MB)
  char* ws = (char*)d_ws;
  _Float16* xh    = (_Float16*)ws; ws += (size_t)ROWS * DIM * 2;
  _Float16* wqkvT = (_Float16*)ws; ws += (size_t)3 * DIM * DIM * 2;
  _Float16* woutT = (_Float16*)ws; ws += (size_t)DIM * DIM * 2;
  _Float16* qm    = (_Float16*)ws; ws += (size_t)ROWS * DIM * 2;
  _Float16* km    = (_Float16*)ws; ws += (size_t)ROWS * DIM * 2;
  _Float16* vtm   = (_Float16*)ws; ws += (size_t)ROWS * DIM * 2;
  _Float16* ohm   = (_Float16*)ws;

  int n1 = ROWS * DIM;
  prep_cvt<<<(n1 + 255) / 256, 256, 0, stream>>>(x, xh, n1);
  int n2 = DIM * 3 * DIM;
  prep_transpose<<<(n2 + 255) / 256, 256, 0, stream>>>(w_qkv, wqkvT, DIM, 3 * DIM);
  int n3 = DIM * DIM;
  prep_transpose<<<(n3 + 255) / 256, 256, 0, stream>>>(w_out, woutT, DIM, DIM);

  gemm_qkv <<<dim3(ROWS / 32, 3 * DIM / 64), 32, 0, stream>>>(xh, wqkvT, b_qkv, qm, km, vtm);
  flash_attn<<<dim3(NTOK / 32, BATCH * HEADS), 32, 0, stream>>>(qm, km, vtm, ohm);
  gemm_out <<<dim3(ROWS / 32, DIM / 64),      32, 0, stream>>>(ohm, woutT, b_out, out);
}
